// social_stgcn_79920751444229
// MI455X (gfx1250) — compile-verified, matrix operand order checked by reading, not run on hardware
//
#include <hip/hip_runtime.h>

typedef __bf16 bf16_t;
typedef __attribute__((ext_vector_type(16))) __bf16 v16bf;
typedef __attribute__((ext_vector_type(8)))  __bf16 bv8;
typedef __attribute__((ext_vector_type(8)))  float  v8f;
typedef __attribute__((ext_vector_type(4)))  float  f4;

#define N_NODES 20000
#define N_EDGES 320000
#define BATCH 4
#define T_IN 12
#define T1V 10
#define T2V 8
#define HIDV 64
#define OUTCV 5
#define FILV 32
#define NGRAPH (BATCH*T1V)
#define BN_EPS 1e-5f

// ---------------- utility kernels ----------------
__global__ void k_zero(float* p, long n) {
  long i = (long)blockIdx.x * blockDim.x + threadIdx.x;
  if (i < n) p[i] = 0.f;
}

// Pack f32 row-major weights (nmats of [kchunks*32 x ncols]) into the per-lane
// WMMA B-operand image: out[((mat*kchunks+kc)*ntiles+nt)*32 + lane][0..15].
__global__ void k_pack_wmma_b(const float* __restrict__ W, bf16_t* __restrict__ out,
                              int ncols, int kchunks, int ntiles, int nmats) {
  int id = blockIdx.x * blockDim.x + threadIdx.x;
  int total = nmats * kchunks * ntiles * 32;
  if (id >= total) return;
  int lane = id & 31;
  int r = id >> 5;
  int nt = r % ntiles; r /= ntiles;
  int kc = r % kchunks; r /= kchunks;
  int mat = r;
  int koff = (lane & 16) ? 8 : 0;
  int kb = kc * 32 + koff;
  int col = nt * 16 + (lane & 15);
  const float* Wm = W + (size_t)mat * kchunks * 32 * ncols;
  bf16_t* op = out + (size_t)id * 16;
#pragma unroll
  for (int i = 0; i < 8; i++) {
    op[i]     = (bf16_t)Wm[(kb + i) * ncols + col];
    op[i + 8] = (bf16_t)Wm[(kb + 16 + i) * ncols + col];
  }
}

// ---------------- graph normalization ----------------
__global__ void k_degree(const int* ei, float* deg) {
  int e = blockIdx.x * blockDim.x + threadIdx.x;
  if (e >= N_EDGES) return;
  int s = ei[e], d = ei[N_EDGES + e];
  if (s != d) atomicAdd(&deg[s], 1.f);
}

__global__ void k_dinv(float* deg) {
  int n = blockIdx.x * blockDim.x + threadIdx.x;
  if (n >= N_NODES) return;
  float v = deg[n];
  deg[n] = (v > 0.f) ? rsqrtf(v) : 0.f;
}

__global__ void k_ew(const int* ei, const float* dinv, float* ew) {
  int e = blockIdx.x * blockDim.x + threadIdx.x;
  if (e >= N_EDGES) return;
  int s = ei[e], d = ei[N_EDGES + e];
  ew[e] = (s != d) ? (-dinv[s] * dinv[d]) : 0.f;
}

// ---------------- temporal conv 1 : (B,T,N,2) -> bf16 (B*T1, N, 64) ----------------
__global__ void k_tconv1(const float* __restrict__ x, const float* __restrict__ w,
                         const float* __restrict__ bias, bf16_t* __restrict__ h1) {
  __shared__ float sw[3 * 3 * 2 * HIDV];
  __shared__ float sb[3 * HIDV];
  for (int i = threadIdx.x; i < 3 * 3 * 2 * HIDV; i += blockDim.x) sw[i] = w[i];
  for (int i = threadIdx.x; i < 3 * HIDV; i += blockDim.x) sb[i] = bias[i];
  __syncthreads();
  long idx = (long)blockIdx.x * blockDim.x + threadIdx.x;
  long total = (long)NGRAPH * N_NODES * HIDV;
  if (idx >= total) return;
  int c = (int)(idx % HIDV);
  long r = idx / HIDV;
  int n = (int)(r % N_NODES);
  int g = (int)(r / N_NODES);
  int b = g / T1V, t = g % T1V;
  float accP = sb[0 * HIDV + c], accQ = sb[1 * HIDV + c], accC = sb[2 * HIDV + c];
#pragma unroll
  for (int k = 0; k < 3; k++) {
    const float* xp = x + ((long)(b * T_IN + t + k) * N_NODES + n) * 2;
#pragma unroll
    for (int ci = 0; ci < 2; ci++) {
      float xv = xp[ci];
      accP += xv * sw[((0 * 3 + k) * 2 + ci) * HIDV + c];
      accQ += xv * sw[((1 * 3 + k) * 2 + ci) * HIDV + c];
      accC += xv * sw[((2 * 3 + k) * 2 + ci) * HIDV + c];
    }
  }
  float q = 1.f / (1.f + __expf(-accQ));
  float v = accP * q + accC;
  h1[idx] = (bf16_t)(v > 0.f ? v : 0.f);
}

// ---------------- scatter (sparse propagation) ----------------
__global__ void k_scatter_bf16(float* __restrict__ out, const bf16_t* __restrict__ in,
                               const float* __restrict__ ew, const int* __restrict__ ei) {
  long id = (long)blockIdx.x * blockDim.x + threadIdx.x;
  long total = (long)N_EDGES * 8;
  if (id >= total) return;
  int e = (int)(id >> 3);
  int ch = ((int)id & 7) * 8;
  float w = ew[e];
  int s = ei[e], d = ei[N_EDGES + e];
  bv8 v = *(const bv8*)(in + (long)s * HIDV + ch);   // 16B aligned vector gather
  float* op = out + (long)d * HIDV + ch;
#pragma unroll
  for (int i = 0; i < 8; i++) atomicAdd(op + i, w * (float)v[i]);
}

__global__ void k_scatter_f32(float* __restrict__ out, const float* __restrict__ in,
                              const float* __restrict__ ew, const int* __restrict__ ei,
                              int C, int CW, int nchunk, long sOutB, long sInB) {
  int b = blockIdx.y;
  long id = (long)blockIdx.x * blockDim.x + threadIdx.x;
  long total = (long)N_EDGES * nchunk;
  if (id >= total) return;
  int e = (int)(id / nchunk);
  int ch = (int)(id % nchunk) * CW;
  float w = ew[e];
  int s = ei[e], d = ei[N_EDGES + e];
  const float* ip = in + (long)b * sInB + (long)s * C;
  float* op = out + (long)b * sOutB + (long)d * C;
  int cend = ch + CW; if (cend > C) cend = C;
  for (int c = ch; c < cend; ++c) atomicAdd(op + c, w * ip[c]);
}

// ---------------- Chebyshev GEMM (one graph): out = relu(Tx0 W0 + Tx1 W1 + (2 P2 - Tx0) W2 + b)
// bf16 WMMA, f32 accumulate; B operands pre-packed per-lane (32B vector loads).
__global__ void k_chebgemm(const bf16_t* __restrict__ tx0, const float* __restrict__ p1,
                           const float* __restrict__ p2, const v16bf* __restrict__ wp,
                           const float* __restrict__ bias, bf16_t* __restrict__ out) {
  int lane = threadIdx.x & 31;
  int wave = threadIdx.x >> 5;
  int tile = blockIdx.x * 4 + wave;               // 128 threads = 4 waves
  if (tile >= N_NODES / 16) return;               // wave-uniform exit
  int mrow = tile * 16 + (lane & 15);
  int koff = (lane & 16) ? 8 : 0;

  const bf16_t* r0 = tx0 + (long)mrow * HIDV;
  const float*  r1 = p1 + (long)mrow * HIDV;
  const float*  r2 = p2 + (long)mrow * HIDV;

  v16bf a0[2], a1[2], a2[2];
#pragma unroll
  for (int kc = 0; kc < 2; kc++) {
    int kb = kc * 32 + koff;
    bv8 t0lo = *(const bv8*)(r0 + kb);
    bv8 t0hi = *(const bv8*)(r0 + kb + 16);
    f4 p1a = *(const f4*)(r1 + kb);      f4 p1b = *(const f4*)(r1 + kb + 4);
    f4 p1c = *(const f4*)(r1 + kb + 16); f4 p1d = *(const f4*)(r1 + kb + 20);
    f4 p2a = *(const f4*)(r2 + kb);      f4 p2b = *(const f4*)(r2 + kb + 4);
    f4 p2c = *(const f4*)(r2 + kb + 16); f4 p2d = *(const f4*)(r2 + kb + 20);
#pragma unroll
    for (int i = 0; i < 4; i++) {
      a0[kc][i]      = t0lo[i];
      a0[kc][i + 4]  = t0lo[i + 4];
      a0[kc][i + 8]  = t0hi[i];
      a0[kc][i + 12] = t0hi[i + 4];
      a1[kc][i]      = (bf16_t)p1a[i];
      a1[kc][i + 4]  = (bf16_t)p1b[i];
      a1[kc][i + 8]  = (bf16_t)p1c[i];
      a1[kc][i + 12] = (bf16_t)p1d[i];
      a2[kc][i]      = (bf16_t)(2.f * p2a[i] - (float)t0lo[i]);
      a2[kc][i + 4]  = (bf16_t)(2.f * p2b[i] - (float)t0lo[i + 4]);
      a2[kc][i + 8]  = (bf16_t)(2.f * p2c[i] - (float)t0hi[i]);
      a2[kc][i + 12] = (bf16_t)(2.f * p2d[i] - (float)t0hi[i + 4]);
    }
  }
  int col15 = lane & 15;
  v8f acc[4];
#pragma unroll
  for (int nt = 0; nt < 4; nt++)
#pragma unroll
    for (int i = 0; i < 8; i++) acc[nt][i] = 0.f;

#pragma unroll
  for (int nt = 0; nt < 4; nt++) {
#pragma unroll
    for (int mat = 0; mat < 3; mat++) {
#pragma unroll
      for (int kc = 0; kc < 2; kc++) {
        v16bf bt = wp[(((mat * 2 + kc) * 4 + nt) << 5) + lane];  // 32B vector load
        v16bf A = (mat == 0) ? a0[kc] : (mat == 1) ? a1[kc] : a2[kc];
        acc[nt] = __builtin_amdgcn_wmma_f32_16x16x32_bf16(
            false, A, false, bt, (short)0, acc[nt], false, false);
      }
    }
  }
  int mbase = tile * 16 + ((lane & 16) ? 8 : 0);
#pragma unroll
  for (int nt = 0; nt < 4; nt++) {
    int col = nt * 16 + col15;
    float bv = bias[col];
#pragma unroll
    for (int i = 0; i < 8; i++) {
      float v = acc[nt][i] + bv;
      out[(long)(mbase + i) * HIDV + col] = (bf16_t)(v > 0.f ? v : 0.f);
    }
  }
}

// ---------------- temporal conv 2 : bf16 (B*T1,N,64) -> f32 (B,T2,N,5) ----------------
__global__ void k_tconv2(const bf16_t* __restrict__ hc, const float* __restrict__ w,
                         const float* __restrict__ bias, float* __restrict__ h2) {
  __shared__ float sw[3 * 3 * HIDV * OUTCV];  // 2880 floats
  __shared__ float sb[3 * OUTCV];
  for (int i = threadIdx.x; i < 3 * 3 * HIDV * OUTCV; i += blockDim.x) sw[i] = w[i];
  for (int i = threadIdx.x; i < 3 * OUTCV; i += blockDim.x) sb[i] = bias[i];
  __syncthreads();
  int idx = blockIdx.x * blockDim.x + threadIdx.x;
  if (idx >= BATCH * T2V * N_NODES) return;
  int n = idx % N_NODES;
  int r = idx / N_NODES;
  int t = r % T2V, b = r / T2V;
  float aP[OUTCV], aQ[OUTCV], aC[OUTCV];
#pragma unroll
  for (int co = 0; co < OUTCV; co++) {
    aP[co] = sb[0 * OUTCV + co]; aQ[co] = sb[1 * OUTCV + co]; aC[co] = sb[2 * OUTCV + co];
  }
#pragma unroll
  for (int k = 0; k < 3; k++) {
    const bf16_t* row = hc + ((long)((b * T1V + t + k) * N_NODES + n)) * HIDV;
    for (int ci8 = 0; ci8 < HIDV; ci8 += 8) {
      bv8 vv = *(const bv8*)(row + ci8);
#pragma unroll
      for (int j = 0; j < 8; j++) {
        float v = (float)vv[j];
        int ci = ci8 + j;
#pragma unroll
        for (int co = 0; co < OUTCV; co++) {
          aP[co] += v * sw[((0 * 3 + k) * HIDV + ci) * OUTCV + co];
          aQ[co] += v * sw[((1 * 3 + k) * HIDV + ci) * OUTCV + co];
          aC[co] += v * sw[((2 * 3 + k) * HIDV + ci) * OUTCV + co];
        }
      }
    }
  }
  float* op = h2 + (long)idx * OUTCV;
#pragma unroll
  for (int co = 0; co < OUTCV; co++) {
    float q = 1.f / (1.f + __expf(-aQ[co]));
    float v = aP[co] * q + aC[co];
    op[co] = (v > 0.f ? v : 0.f);
  }
}

// ---------------- BatchNorm per-node (train stats) ----------------
__global__ void k_bnstat(const float* __restrict__ h2, float* __restrict__ bnm,
                         float* __restrict__ bnr) {
  int n = blockIdx.x * blockDim.x + threadIdx.x;
  if (n >= N_NODES) return;
  float s = 0.f, s2 = 0.f;
  for (int b = 0; b < BATCH; b++)
    for (int t = 0; t < T2V; t++) {
      const float* p = h2 + ((long)((b * T2V + t) * N_NODES + n)) * OUTCV;
#pragma unroll
      for (int c = 0; c < OUTCV; c++) { float v = p[c]; s += v; s2 += v * v; }
    }
  const float inv = 1.f / (BATCH * T2V * OUTCV);
  float m = s * inv;
  float var = s2 * inv - m * m;
  bnm[n] = m;
  bnr[n] = rsqrtf(var + BN_EPS);
}

__global__ void k_bnapply(const float* __restrict__ h2, const float* __restrict__ bnm,
                          const float* __restrict__ bnr, const float* __restrict__ gamma,
                          const float* __restrict__ beta, float* __restrict__ xg) {
  long idx = (long)blockIdx.x * blockDim.x + threadIdx.x;
  long total = (long)BATCH * T2V * N_NODES * OUTCV;
  if (idx >= total) return;
  int n = (int)((idx / OUTCV) % N_NODES);
  float v = (h2[idx] - bnm[n]) * bnr[n] * gamma[n] + beta[n];
  xg[idx] = (v > 0.f ? v : 0.f);
}

// ---------------- GRU gates Z,R (bf16 WMMA over Wh, K=32) ----------------
__global__ void k_gru_zr(const float* __restrict__ Hp, const float* __restrict__ PH,
                         const float* __restrict__ XG, const float* __restrict__ PX,
                         const v16bf* __restrict__ wph, const float* __restrict__ wx,
                         const float* __restrict__ bx, const float* __restrict__ bh,
                         float* __restrict__ Z, float* __restrict__ HR, int tstep) {
  int b = blockIdx.y;
  int lane = threadIdx.x & 31;
  int wave = threadIdx.x >> 5;
  int tile = blockIdx.x * 4 + wave;
  if (tile >= N_NODES / 16) return;
  int mrow = tile * 16 + (lane & 15);
  int koff = (lane & 16) ? 8 : 0;
  const float* hp = Hp + (long)b * N_NODES * FILV;
  const float* ph = PH + (long)b * N_NODES * FILV;
  const float* hrow = hp + (long)mrow * FILV + koff;
  const float* prow = ph + (long)mrow * FILV + koff;
  f4 h0 = *(const f4*)(hrow);      f4 h1v = *(const f4*)(hrow + 4);
  f4 h2v = *(const f4*)(hrow + 16); f4 h3 = *(const f4*)(hrow + 20);
  f4 q0 = *(const f4*)(prow);      f4 q1 = *(const f4*)(prow + 4);
  f4 q2 = *(const f4*)(prow + 16); f4 q3 = *(const f4*)(prow + 20);
  v16bf ah, ap;
#pragma unroll
  for (int i = 0; i < 4; i++) {
    ah[i] = (bf16_t)h0[i]; ah[i + 4] = (bf16_t)h1v[i];
    ah[i + 8] = (bf16_t)h2v[i]; ah[i + 12] = (bf16_t)h3[i];
    ap[i] = (bf16_t)q0[i]; ap[i + 4] = (bf16_t)q1[i];
    ap[i + 8] = (bf16_t)q2[i]; ap[i + 12] = (bf16_t)q3[i];
  }
  int col15 = lane & 15;
  int mbase = tile * 16 + ((lane & 16) ? 8 : 0);
  const float* xgrow = XG + ((long)(b * T2V + tstep) * N_NODES) * OUTCV;
  const float* pxrow = PX + (long)b * N_NODES * OUTCV;
#pragma unroll
  for (int gate = 0; gate < 2; gate++) {
#pragma unroll
    for (int nt = 0; nt < 2; nt++) {
      int col = nt * 16 + col15;
      v16bf b0 = wph[((((gate * 2 + 0)) * 2 + nt) << 5) + lane];
      v16bf b1 = wph[((((gate * 2 + 1)) * 2 + nt) << 5) + lane];
      v8f acc;
#pragma unroll
      for (int i = 0; i < 8; i++) acc[i] = 0.f;
      acc = __builtin_amdgcn_wmma_f32_16x16x32_bf16(false, ah, false, b0, (short)0, acc, false, false);
      acc = __builtin_amdgcn_wmma_f32_16x16x32_bf16(false, ap, false, b1, (short)0, acc, false, false);
#pragma unroll
      for (int i = 0; i < 8; i++) {
        int m = mbase + i;
        float s = bx[gate * FILV + col] + bh[gate * FILV + col];
        const float* xr = xgrow + (long)m * OUTCV;
        const float* pr = pxrow + (long)m * OUTCV;
#pragma unroll
        for (int c = 0; c < OUTCV; c++) {
          s += xr[c] * wx[((gate * 2 + 0) * OUTCV + c) * FILV + col];
          s += pr[c] * wx[((gate * 2 + 1) * OUTCV + c) * FILV + col];
        }
        float g = 1.f / (1.f + __expf(-(acc[i] + s)));
        long o = (long)b * N_NODES * FILV + (long)m * FILV + col;
        if (gate == 0) Z[o] = g;
        else HR[o] = hp[(long)m * FILV + col] * g;
      }
    }
  }
}

// ---------------- GRU candidate + combine ----------------
__global__ void k_gru_h(const float* __restrict__ HRb, const float* __restrict__ PHR,
                        const float* __restrict__ XG, const float* __restrict__ PX,
                        const v16bf* __restrict__ wph, const float* __restrict__ wx,
                        const float* __restrict__ bx, const float* __restrict__ bh,
                        const float* __restrict__ Z, const float* __restrict__ Hp,
                        float* __restrict__ Hn, int tstep) {
  int b = blockIdx.y;
  int lane = threadIdx.x & 31;
  int wave = threadIdx.x >> 5;
  int tile = blockIdx.x * 4 + wave;
  if (tile >= N_NODES / 16) return;
  int mrow = tile * 16 + (lane & 15);
  int koff = (lane & 16) ? 8 : 0;
  const float* hr = HRb + (long)b * N_NODES * FILV;
  const float* pr2 = PHR + (long)b * N_NODES * FILV;
  const float* hrow = hr + (long)mrow * FILV + koff;
  const float* prow = pr2 + (long)mrow * FILV + koff;
  f4 h0 = *(const f4*)(hrow);      f4 h1v = *(const f4*)(hrow + 4);
  f4 h2v = *(const f4*)(hrow + 16); f4 h3 = *(const f4*)(hrow + 20);
  f4 q0 = *(const f4*)(prow);      f4 q1 = *(const f4*)(prow + 4);
  f4 q2 = *(const f4*)(prow + 16); f4 q3 = *(const f4*)(prow + 20);
  v16bf ah, ap;
#pragma unroll
  for (int i = 0; i < 4; i++) {
    ah[i] = (bf16_t)h0[i]; ah[i + 4] = (bf16_t)h1v[i];
    ah[i + 8] = (bf16_t)h2v[i]; ah[i + 12] = (bf16_t)h3[i];
    ap[i] = (bf16_t)q0[i]; ap[i + 4] = (bf16_t)q1[i];
    ap[i + 8] = (bf16_t)q2[i]; ap[i + 12] = (bf16_t)q3[i];
  }
  int col15 = lane & 15;
  int mbase = tile * 16 + ((lane & 16) ? 8 : 0);
  const float* xgrow = XG + ((long)(b * T2V + tstep) * N_NODES) * OUTCV;
  const float* pxrow = PX + (long)b * N_NODES * OUTCV;
#pragma unroll
  for (int nt = 0; nt < 2; nt++) {
    int col = nt * 16 + col15;
    v16bf b0 = wph[((((2 * 2 + 0)) * 2 + nt) << 5) + lane];
    v16bf b1 = wph[((((2 * 2 + 1)) * 2 + nt) << 5) + lane];
    v8f acc;
#pragma unroll
    for (int i = 0; i < 8; i++) acc[i] = 0.f;
    acc = __builtin_amdgcn_wmma_f32_16x16x32_bf16(false, ah, false, b0, (short)0, acc, false, false);
    acc = __builtin_amdgcn_wmma_f32_16x16x32_bf16(false, ap, false, b1, (short)0, acc, false, false);
#pragma unroll
    for (int i = 0; i < 8; i++) {
      int m = mbase + i;
      float s = bx[2 * FILV + col] + bh[2 * FILV + col];
      const float* xr = xgrow + (long)m * OUTCV;
      const float* pxp = pxrow + (long)m * OUTCV;
#pragma unroll
      for (int c = 0; c < OUTCV; c++) {
        s += xr[c] * wx[((2 * 2 + 0) * OUTCV + c) * FILV + col];
        s += pxp[c] * wx[((2 * 2 + 1) * OUTCV + c) * FILV + col];
      }
      float ht = tanhf(acc[i] + s);
      long o = (long)b * N_NODES * FILV + (long)m * FILV + col;
      float z = Z[o];
      float hpv = Hp[o];
      Hn[o] = z * hpv + (1.f - z) * ht;
    }
  }
}

// ---------------- per-step score + final log-softmax over time ----------------
__global__ void k_score(const float* __restrict__ Hn, const float* __restrict__ lw,
                        const float* __restrict__ lb, float* __restrict__ S, int tstep) {
  int idx = blockIdx.x * blockDim.x + threadIdx.x;
  if (idx >= BATCH * N_NODES) return;
  int b = idx / N_NODES, n = idx % N_NODES;
  const float* h = Hn + ((long)b * N_NODES + n) * FILV;
  float s = lb[0];
#pragma unroll
  for (int c = 0; c < FILV; c++) {
    float v = h[c];
    v = v > 0.f ? v : 0.f;
    s += v * lw[c];
  }
  S[((long)b * T2V + tstep) * N_NODES + n] = s;
}

__global__ void k_logsoftmax(const float* __restrict__ S, float* __restrict__ out) {
  int idx = blockIdx.x * blockDim.x + threadIdx.x;
  if (idx >= BATCH * N_NODES) return;
  int b = idx / N_NODES, n = idx % N_NODES;
  float vals[T2V];
  float m = -3.4e38f;
#pragma unroll
  for (int t = 0; t < T2V; t++) {
    float v = S[((long)b * T2V + t) * N_NODES + n];
    vals[t] = v;
    m = v > m ? v : m;
  }
  float l = 0.f;
#pragma unroll
  for (int t = 0; t < T2V; t++) l += __expf(vals[t] - m);
  l = __logf(l);
#pragma unroll
  for (int t = 0; t < T2V; t++)
    out[((long)b * T2V + t) * N_NODES + n] = vals[t] - m - l;
}

// ---------------- host orchestration ----------------
static inline unsigned gblk(long n, int b) { return (unsigned)((n + b - 1) / b); }

extern "C" void kernel_launch(void* const* d_in, const int* in_sizes, int n_in,
                              void* d_out, int out_size, void* d_ws, size_t ws_size,
                              hipStream_t stream) {
  const float* x      = (const float*)d_in[0];
  const int*   ei     = (const int*)d_in[1];
  const float* tc1w   = (const float*)d_in[2];
  const float* tc1b   = (const float*)d_in[3];
  const float* chebw  = (const float*)d_in[4];
  const float* chebb  = (const float*)d_in[5];
  const float* tc2w   = (const float*)d_in[6];
  const float* tc2b   = (const float*)d_in[7];
  const float* bng    = (const float*)d_in[8];
  const float* bnb    = (const float*)d_in[9];
  const float* gwx    = (const float*)d_in[10];
  const float* gbx    = (const float*)d_in[11];
  const float* gwh    = (const float*)d_in[12];
  const float* gbh    = (const float*)d_in[13];
  const float* linw   = (const float*)d_in[14];
  const float* linb   = (const float*)d_in[15];
  float* out = (float*)d_out;

  // workspace carve-up (256B aligned)
  char* w = (char*)d_ws;
  size_t off = 0;
  auto alloc = [&](size_t bytes) { void* p = w + off; off = (off + bytes + 255) & ~(size_t)255; return p; };
  float*  deg = (float*)alloc((size_t)N_NODES * 4);
  float*  ew  = (float*)alloc((size_t)N_EDGES * 4);
  bf16_t* h1  = (bf16_t*)alloc((size_t)NGRAPH * N_NODES * HIDV * 2);
  bf16_t* hc  = (bf16_t*)alloc((size_t)NGRAPH * N_NODES * HIDV * 2);
  float*  P1  = (float*)alloc((size_t)N_NODES * HIDV * 4);
  float*  P2  = (float*)alloc((size_t)N_NODES * HIDV * 4);
  float*  h2  = (float*)alloc((size_t)BATCH * T2V * N_NODES * OUTCV * 4);
  float*  bnm = (float*)alloc((size_t)N_NODES * 4);
  float*  bnr = (float*)alloc((size_t)N_NODES * 4);
  float*  XG  = (float*)alloc((size_t)BATCH * T2V * N_NODES * OUTCV * 4);
  bf16_t* WcP = (bf16_t*)alloc((size_t)3 * 2 * 4 * 32 * 16 * 2);   // packed cheb B
  bf16_t* WhP = (bf16_t*)alloc((size_t)6 * 1 * 2 * 32 * 16 * 2);   // packed GRU Wh B
  float*  PX  = (float*)alloc((size_t)BATCH * N_NODES * OUTCV * 4);
  float*  PH  = (float*)alloc((size_t)BATCH * N_NODES * FILV * 4);
  float*  PHR = (float*)alloc((size_t)BATCH * N_NODES * FILV * 4);
  float*  Zg  = (float*)alloc((size_t)BATCH * N_NODES * FILV * 4);
  float*  HRb = (float*)alloc((size_t)BATCH * N_NODES * FILV * 4);
  float*  HA  = (float*)alloc((size_t)BATCH * N_NODES * FILV * 4);
  float*  HB  = (float*)alloc((size_t)BATCH * N_NODES * FILV * 4);
  float*  S   = (float*)alloc((size_t)BATCH * T2V * N_NODES * 4);
  (void)ws_size; (void)in_sizes; (void)n_in; (void)out_size;

  const int TB = 256;

  // edge weights
  k_zero<<<gblk(N_NODES, TB), TB, 0, stream>>>(deg, N_NODES);
  k_degree<<<gblk(N_EDGES, TB), TB, 0, stream>>>(ei, deg);
  k_dinv<<<gblk(N_NODES, TB), TB, 0, stream>>>(deg);
  k_ew<<<gblk(N_EDGES, TB), TB, 0, stream>>>(ei, deg, ew);

  // pack weights into per-lane WMMA B operand images
  k_pack_wmma_b<<<gblk(3 * 2 * 4 * 32, TB), TB, 0, stream>>>(chebw, WcP, HIDV, 2, 4, 3);
  k_pack_wmma_b<<<gblk(6 * 1 * 2 * 32, TB), TB, 0, stream>>>(gwh, WhP, FILV, 1, 2, 6);

  // temporal conv 1
  {
    long total = (long)NGRAPH * N_NODES * HIDV;
    k_tconv1<<<gblk(total, TB), TB, 0, stream>>>(x, tc1w, tc1b, h1);
  }

  // Chebyshev graph conv per graph (Tx0 in h1, out in hc)
  const int tilesN = N_NODES / 16;
  const dim3 wblk(128);
  const dim3 wgrid((tilesN + 3) / 4);
  for (int g = 0; g < NGRAPH; g++) {
    const bf16_t* tx0 = h1 + (long)g * N_NODES * HIDV;
    bf16_t* outg = hc + (long)g * N_NODES * HIDV;
    k_zero<<<gblk((long)N_NODES * HIDV, TB), TB, 0, stream>>>(P1, (long)N_NODES * HIDV);
    k_scatter_bf16<<<gblk((long)N_EDGES * 8, TB), TB, 0, stream>>>(P1, tx0, ew, ei);
    k_zero<<<gblk((long)N_NODES * HIDV, TB), TB, 0, stream>>>(P2, (long)N_NODES * HIDV);
    k_scatter_f32<<<dim3(gblk((long)N_EDGES * 8, TB), 1), TB, 0, stream>>>(
        P2, P1, ew, ei, HIDV, 8, 8, 0, 0);
    k_chebgemm<<<wgrid, wblk, 0, stream>>>(tx0, P1, P2, (const v16bf*)WcP, chebb, outg);
  }

  // temporal conv 2 + batchnorm + relu
  k_tconv2<<<gblk((long)BATCH * T2V * N_NODES, TB), TB, 0, stream>>>(hc, tc2w, tc2b, h2);
  k_bnstat<<<gblk(N_NODES, TB), TB, 0, stream>>>(h2, bnm, bnr);
  k_bnapply<<<gblk((long)BATCH * T2V * N_NODES * OUTCV, TB), TB, 0, stream>>>(
      h2, bnm, bnr, bng, bnb, XG);

  // GConvGRU over T2 steps
  k_zero<<<gblk((long)BATCH * N_NODES * FILV, TB), TB, 0, stream>>>(HA, (long)BATCH * N_NODES * FILV);
  float* Hcur = HA;
  float* Hnxt = HB;
  for (int t = 0; t < T2V; t++) {
    k_zero<<<gblk((long)BATCH * N_NODES * OUTCV, TB), TB, 0, stream>>>(PX, (long)BATCH * N_NODES * OUTCV);
    k_zero<<<gblk((long)BATCH * N_NODES * FILV, TB), TB, 0, stream>>>(PH, (long)BATCH * N_NODES * FILV);
    // PX = prop(X_t) per batch
    k_scatter_f32<<<dim3(gblk((long)N_EDGES, TB), BATCH), TB, 0, stream>>>(
        PX, XG + (long)t * N_NODES * OUTCV, ew, ei, OUTCV, OUTCV, 1,
        (long)N_NODES * OUTCV, (long)T2V * N_NODES * OUTCV);
    // PH = prop(H) per batch
    k_scatter_f32<<<dim3(gblk((long)N_EDGES * 4, TB), BATCH), TB, 0, stream>>>(
        PH, Hcur, ew, ei, FILV, 8, 4, (long)N_NODES * FILV, (long)N_NODES * FILV);
    // Z, R gates (+ HR = H*R)
    k_gru_zr<<<dim3(wgrid.x, BATCH), wblk, 0, stream>>>(
        Hcur, PH, XG, PX, (const v16bf*)WhP, gwx, gbx, gbh, Zg, HRb, t);
    // PHR = prop(H*R)
    k_zero<<<gblk((long)BATCH * N_NODES * FILV, TB), TB, 0, stream>>>(PHR, (long)BATCH * N_NODES * FILV);
    k_scatter_f32<<<dim3(gblk((long)N_EDGES * 4, TB), BATCH), TB, 0, stream>>>(
        PHR, HRb, ew, ei, FILV, 8, 4, (long)N_NODES * FILV, (long)N_NODES * FILV);
    // candidate + combine
    k_gru_h<<<dim3(wgrid.x, BATCH), wblk, 0, stream>>>(
        HRb, PHR, XG, PX, (const v16bf*)WhP, gwx, gbx, gbh, Zg, Hcur, Hnxt, t);
    // score for this step
    k_score<<<gblk((long)BATCH * N_NODES, TB), TB, 0, stream>>>(Hnxt, linw, linb, S, t);
    float* tmp = Hcur; Hcur = Hnxt; Hnxt = tmp;
  }

  // log-softmax over time axis
  k_logsoftmax<<<gblk((long)BATCH * N_NODES, TB), TB, 0, stream>>>(S, out);
}
